// GCN_22892175687930
// MI455X (gfx1250) — compile-verified, matrix operand order checked by reading, not run on hardware
//
#include <hip/hip_runtime.h>
#include <hip/hip_bf16.h>

// ---------------- constants ----------------
#define NN   50000
#define EE   400000
#define DD   200
#define HH   10
#define DKK  20
#define BB   50
#define NPGc 1000
#define ETc  6
#define L1c  1000
#define L2c  333
#define L3c  111
#define EPSc 1e-5f

typedef __attribute__((ext_vector_type(16))) __bf16 v16bf;
typedef __attribute__((ext_vector_type(8)))  __bf16 v8bf;
typedef __attribute__((ext_vector_type(8)))  float  v8f;

// ---------------- operand packing ----------------
// dst (bf16) is Rp x Kp row-major, zero padded.
// transpose==0: dst[r][k] = src[r*lds + k]
// transpose==1: dst[r][k] = src[k*lds + r]   (packs B as N x K)
__global__ void pack_bf16(const float* __restrict__ src, __bf16* __restrict__ dst,
                          int R, int K, int lds, int Kp, long long totPerBatch,
                          long long sSrc, long long sDst, int transpose)
{
    long long i = (long long)blockIdx.x * blockDim.x + threadIdx.x;
    if (i >= totPerBatch) return;
    int bz = blockIdx.z;
    src += (long long)bz * sSrc;
    dst += (long long)bz * sDst;
    int k = (int)(i % Kp);
    int r = (int)(i / Kp);
    float v = 0.0f;
    if (r < R && k < K)
        v = transpose ? src[(long long)k * lds + r] : src[(long long)r * lds + k];
    dst[i] = (__bf16)v;
}

// ---------------- WMMA GEMM on packed bf16 ----------------
// C[b](MxN) = act( Ap(MpxKp) * Btp[b](NpxKp)^T + bias (+ res[b]) )
// block tile 128x64, 8 waves, wave tile 32x32 (4 x v_wmma_f32_16x16x32_bf16)
// staging: double-buffered GLOBAL_LOAD_ASYNC_TO_LDS_B128 (ASYNCcnt)
#define TM 128
#define TN 64
#define TK 32
#define LDSP (TK + 8)   // 40 bf16 = 80B row stride (16B aligned, conflict-free)

__device__ __forceinline__ void async_ld16(unsigned ldsOff, const void* gptr)
{
    asm volatile("global_load_async_to_lds_b128 %0, %1, off"
                 :: "v"(ldsOff), "v"(gptr) : "memory");
}

__device__ __forceinline__ void wait_async0()
{
    asm volatile("s_wait_asynccnt 0" ::: "memory");
}

__global__ __launch_bounds__(256) void gemm_wmma(
    const __bf16* __restrict__ Ap, const __bf16* __restrict__ Btp,
    const float* __restrict__ bias, const float* __restrict__ res,
    float* __restrict__ C, int M, int Ncol, int Kp, int ldc,
    long long sBt, long long sR, long long sC, int act, int biasMode)
{
    __shared__ __bf16 As[2][TM][LDSP];   // 2 x 10240 B
    __shared__ __bf16 Bs[2][TN][LDSP];   // 2 x  5120 B

    int bz = blockIdx.z;
    Btp += (long long)bz * sBt;
    C   += (long long)bz * sC;
    const float* resb = res ? res + (long long)bz * sR : nullptr;

    int m0 = blockIdx.y * TM;
    int n0 = blockIdx.x * TN;
    int tid  = threadIdx.x;
    int lane = tid & 31;
    int wave = tid >> 5;
    int wm   = wave >> 1;     // 0..3 : rows [wm*32, +32)
    int wn   = wave & 1;      // 0..1 : cols [wn*32, +32)

    v8f c00 = {}, c01 = {}, c10 = {}, c11 = {};

    // staging indices (16B chunks)
    int ar0 = (tid)       >> 2, as0 = (tid)       & 3;   // A chunk 0
    int ar1 = (tid + 256) >> 2, as1 = (tid + 256) & 3;   // A chunk 1
    int brw = tid >> 2,         bsb = tid & 3;           // B chunk

    // prologue: stage tile 0 into buffer 0
    async_ld16((unsigned)(unsigned long long)&As[0][ar0][as0 * 8],
               &Ap[(long long)(m0 + ar0) * Kp + as0 * 8]);
    async_ld16((unsigned)(unsigned long long)&As[0][ar1][as1 * 8],
               &Ap[(long long)(m0 + ar1) * Kp + as1 * 8]);
    async_ld16((unsigned)(unsigned long long)&Bs[0][brw][bsb * 8],
               &Btp[(long long)(n0 + brw) * Kp + bsb * 8]);
    wait_async0();
    __syncthreads();

    int nTiles = Kp / TK;
    for (int t = 0; t < nTiles; ++t) {
        int cur = t & 1;
        int nxt = cur ^ 1;
        if (t + 1 < nTiles) {
            int k1 = (t + 1) * TK;
            async_ld16((unsigned)(unsigned long long)&As[nxt][ar0][as0 * 8],
                       &Ap[(long long)(m0 + ar0) * Kp + k1 + as0 * 8]);
            async_ld16((unsigned)(unsigned long long)&As[nxt][ar1][as1 * 8],
                       &Ap[(long long)(m0 + ar1) * Kp + k1 + as1 * 8]);
            async_ld16((unsigned)(unsigned long long)&Bs[nxt][brw][bsb * 8],
                       &Btp[(long long)(n0 + brw) * Kp + k1 + bsb * 8]);
        }

        // fragments: k = 8h+s (s<8) / 16+8h+(s-8) (s>=8), two 16B LDS loads each
        int fm   = lane & 15;
        int koff = (lane >> 4) << 3;   // 0 or 8
        v8bf t0, t1;

        t0 = *(const v8bf*)&As[cur][wm * 32 + fm][koff];
        t1 = *(const v8bf*)&As[cur][wm * 32 + fm][koff + 16];
        v16bf a0 = __builtin_shufflevector(t0, t1, 0,1,2,3,4,5,6,7,8,9,10,11,12,13,14,15);
        t0 = *(const v8bf*)&As[cur][wm * 32 + 16 + fm][koff];
        t1 = *(const v8bf*)&As[cur][wm * 32 + 16 + fm][koff + 16];
        v16bf a1 = __builtin_shufflevector(t0, t1, 0,1,2,3,4,5,6,7,8,9,10,11,12,13,14,15);
        t0 = *(const v8bf*)&Bs[cur][wn * 32 + fm][koff];
        t1 = *(const v8bf*)&Bs[cur][wn * 32 + fm][koff + 16];
        v16bf b0 = __builtin_shufflevector(t0, t1, 0,1,2,3,4,5,6,7,8,9,10,11,12,13,14,15);
        t0 = *(const v8bf*)&Bs[cur][wn * 32 + 16 + fm][koff];
        t1 = *(const v8bf*)&Bs[cur][wn * 32 + 16 + fm][koff + 16];
        v16bf b1 = __builtin_shufflevector(t0, t1, 0,1,2,3,4,5,6,7,8,9,10,11,12,13,14,15);

        c00 = __builtin_amdgcn_wmma_f32_16x16x32_bf16(false, a0, false, b0, (short)0, c00, false, false);
        c01 = __builtin_amdgcn_wmma_f32_16x16x32_bf16(false, a0, false, b1, (short)0, c01, false, false);
        c10 = __builtin_amdgcn_wmma_f32_16x16x32_bf16(false, a1, false, b0, (short)0, c10, false, false);
        c11 = __builtin_amdgcn_wmma_f32_16x16x32_bf16(false, a1, false, b1, (short)0, c11, false, false);

        // async loads for tile t+1 overlap the WMMAs above
        wait_async0();
        __syncthreads();
    }

    // epilogue (C/D f32 layout: VGPR r -> M=r (lanes0-15) / M=8+r (lanes16-31))
    int nloc = lane & 15;
    int rofs = (lane >> 4) * 8;
    v8f* accs[4] = { &c00, &c01, &c10, &c11 };
    #pragma unroll
    for (int t = 0; t < 4; ++t) {
        int mb = (t >> 1) * 16;
        int nb = (t & 1) * 16;
        int gn = n0 + wn * 32 + nb + nloc;
        #pragma unroll
        for (int r = 0; r < 8; ++r) {
            int gm = m0 + wm * 32 + mb + rofs + r;
            if (gm < M && gn < Ncol) {
                float vv = (*accs[t])[r];
                if (bias) vv += biasMode ? bias[gm] : bias[gn];
                if (resb) vv += resb[(long long)gm * ldc + gn];
                if (act == 1) vv = fmaxf(vv, 0.0f);
                else if (act == 2) vv = 0.5f * vv * (1.0f + erff(vv * 0.70710678118f));
                C[(long long)gm * ldc + gn] = vv;
            }
        }
    }
}

// ---------------- utility kernels ----------------
__global__ void fill_f32(float* p, float v, long long n) {
    long long i = (long long)blockIdx.x * blockDim.x + threadIdx.x;
    if (i < n) p[i] = v;
}

__device__ __forceinline__ void atomicMaxF(float* addr, float val) {
    unsigned int* ua = (unsigned int*)addr;
    unsigned int old = *ua;
    while (true) {
        float cur = __uint_as_float(old);
        if (cur >= val) break;
        unsigned int assumed = old;
        old = atomicCAS(ua, assumed, __float_as_uint(val));
        if (old == assumed) break;
    }
}

// ---------------- attention kernels ----------------
__global__ void attn_scores(const float* __restrict__ q, const float* __restrict__ k,
                            const int* __restrict__ src, const int* __restrict__ dst,
                            const int* __restrict__ et, const float* __restrict__ ebias,
                            float* __restrict__ s, int E)
{
    int i = blockIdx.x * blockDim.x + threadIdx.x;
    if (i >= E * HH) return;
    int e = i / HH, h = i % HH;
    int dn = dst[e], sn = src[e];
    const float* qp = q + (long long)dn * DD + h * DKK;
    const float* kp = k + (long long)sn * DD + h * DKK;
    float acc = 0.0f;
    #pragma unroll
    for (int j = 0; j < DKK; ++j) acc += qp[j] * kp[j];
    acc = acc * 0.22360679775f + ebias[et[e] * HH + h];
    s[i] = acc;
}

__global__ void attn_max(const float* __restrict__ s, const int* __restrict__ dst,
                         float* __restrict__ m, int E)
{
    int i = blockIdx.x * blockDim.x + threadIdx.x;
    if (i >= E * HH) return;
    int e = i / HH, h = i % HH;
    atomicMaxF(&m[dst[e] * HH + h], s[i]);
}

__global__ void attn_exp(float* __restrict__ s, const int* __restrict__ dst,
                         const float* __restrict__ m, float* __restrict__ z, int E)
{
    int i = blockIdx.x * blockDim.x + threadIdx.x;
    if (i >= E * HH) return;
    int e = i / HH, h = i % HH;
    float ev = __expf(s[i] - m[dst[e] * HH + h]);
    s[i] = ev;
    atomicAdd(&z[dst[e] * HH + h], ev);
}

__global__ void attn_agg(const float* __restrict__ s, const float* __restrict__ v,
                         const int* __restrict__ src, const int* __restrict__ dst,
                         const float* __restrict__ z, float* __restrict__ agg, int E)
{
    int i = blockIdx.x * blockDim.x + threadIdx.x;
    if (i >= E * HH) return;
    int e = i / HH, h = i % HH;
    int dn = dst[e], sn = src[e];
    float w = s[i] / z[dn * HH + h];
    const float* vp = v + (long long)sn * DD + h * DKK;
    float* ap = agg + (long long)dn * DD + h * DKK;
    #pragma unroll
    for (int j = 0; j < DKK; ++j) atomicAdd(&ap[j], w * vp[j]);
}

// ---------------- batchnorm kernels ----------------
__global__ void bn_stats(const float* __restrict__ x, int rows, int cols,
                         float* __restrict__ mean, float* __restrict__ var)
{
    __shared__ float ss[256], sq[256];
    int c = blockIdx.x;
    float s = 0.0f, q = 0.0f;
    for (int r = threadIdx.x; r < rows; r += 256) {
        float v = x[(long long)r * cols + c];
        s += v; q += v * v;
    }
    ss[threadIdx.x] = s; sq[threadIdx.x] = q;
    __syncthreads();
    for (int st = 128; st > 0; st >>= 1) {
        if (threadIdx.x < st) { ss[threadIdx.x] += ss[threadIdx.x + st];
                                sq[threadIdx.x] += sq[threadIdx.x + st]; }
        __syncthreads();
    }
    if (threadIdx.x == 0) {
        float mu = ss[0] / rows;
        mean[c] = mu;
        var[c]  = sq[0] / rows - mu * mu;
    }
}

__global__ void bn_apply(const float* __restrict__ x, const float* __restrict__ mean,
                         const float* __restrict__ var, const float* __restrict__ g,
                         const float* __restrict__ beta, float* __restrict__ out,
                         long long total, int cols)
{
    long long i = (long long)blockIdx.x * blockDim.x + threadIdx.x;
    if (i >= total) return;
    int c = (int)(i % cols);
    out[i] = (x[i] - mean[c]) * rsqrtf(var[c] + EPSc) * g[c] + beta[c];
}

__global__ void bnconv_stats(const float* __restrict__ x, int Bc, int C, int L,
                             float* __restrict__ mean, float* __restrict__ var)
{
    __shared__ float ss[256], sq[256];
    int c = blockIdx.x;
    float s = 0.0f, q = 0.0f;
    int tot = Bc * L;
    for (int t = threadIdx.x; t < tot; t += 256) {
        int b = t / L, l = t % L;
        float v = x[((long long)b * C + c) * L + l];
        s += v; q += v * v;
    }
    ss[threadIdx.x] = s; sq[threadIdx.x] = q;
    __syncthreads();
    for (int st = 128; st > 0; st >>= 1) {
        if (threadIdx.x < st) { ss[threadIdx.x] += ss[threadIdx.x + st];
                                sq[threadIdx.x] += sq[threadIdx.x + st]; }
        __syncthreads();
    }
    if (threadIdx.x == 0) {
        float mu = ss[0] / tot;
        mean[c] = mu;
        var[c]  = sq[0] / tot - mu * mu;
    }
}

__global__ void bnconv_apply(const float* __restrict__ x, const float* __restrict__ mean,
                             const float* __restrict__ var, const float* __restrict__ g,
                             const float* __restrict__ beta, float* __restrict__ out,
                             int total, int L, int C)
{
    int i = blockIdx.x * blockDim.x + threadIdx.x;
    if (i >= total) return;
    int c = (i / L) % C;
    out[i] = (x[i] - mean[c]) * rsqrtf(var[c] + EPSc) * g[c] + beta[c];
}

// ---------------- layout / pooling / readout ----------------
__global__ void to_bcl(const float* __restrict__ h, float* __restrict__ y, int total)
{
    int i = blockIdx.x * blockDim.x + threadIdx.x;
    if (i >= total) return;
    int l = i % NPGc;
    int c = (i / NPGc) % DD;
    int b = i / (NPGc * DD);
    y[i] = h[((long long)b * NPGc + l) * DD + c];
}

__global__ void avgpool3(const float* __restrict__ in, float* __restrict__ out,
                         int Lin, int Lout, int total)
{
    int i = blockIdx.x * blockDim.x + threadIdx.x;
    if (i >= total) return;
    int j  = i % Lout;
    int bc = i / Lout;
    const float* p = in + (long long)bc * Lin + j * 3;
    out[i] = (p[0] + p[1] + p[2]) * (1.0f / 3.0f);
}

__global__ void rowsum(const float* __restrict__ in, float* __restrict__ out,
                       int Lc, int total)
{
    int i = blockIdx.x * blockDim.x + threadIdx.x;
    if (i >= total) return;
    const float* p = in + (long long)i * Lc;
    float s = 0.0f;
    for (int l = 0; l < Lc; ++l) s += p[l];
    out[i] = s;
}

__global__ void softmax2(const float* __restrict__ logits, float* __restrict__ out, int n)
{
    int i = blockIdx.x * blockDim.x + threadIdx.x;
    if (i >= n) return;
    float a = logits[i * 2], b = logits[i * 2 + 1];
    float mx = fmaxf(a, b);
    float ea = __expf(a - mx), eb = __expf(b - mx);
    float inv = 1.0f / (ea + eb);
    out[i * 2]     = ea * inv;
    out[i * 2 + 1] = eb * inv;
}

// ---------------- host orchestration ----------------
static inline int gup(long long n) { return (int)((n + 255) / 256); }
static inline int rup(int x, int m) { return (x + m - 1) / m * m; }

static void run_gemm(hipStream_t stream, float* paBuf, float* pbBuf,
                     const float* A, int lda,
                     const float* Bsrc, int ldbs, long long sBsrc,
                     const float* bias, const float* res,
                     float* C, int M, int Ncol, int K, int ldc,
                     long long sR, long long sC, int batch, int act, int biasMode)
{
    int Kp = rup(K, 32);
    int Mp = rup(M, 128);
    int Np = rup(Ncol, 64);
    __bf16* Ap = (__bf16*)paBuf;
    __bf16* Bt = (__bf16*)pbBuf;
    long long aTot = (long long)Mp * Kp;
    long long bTot = (long long)Np * Kp;
    pack_bf16<<<dim3(gup(aTot), 1, 1), 256, 0, stream>>>(
        A, Ap, M, K, lda, Kp, aTot, 0, 0, 0);
    pack_bf16<<<dim3(gup(bTot), 1, batch), 256, 0, stream>>>(
        Bsrc, Bt, Ncol, K, ldbs, Kp, bTot, sBsrc, bTot, 1);
    dim3 grid(Np / 64, Mp / 128, batch);
    gemm_wmma<<<grid, dim3(256), 0, stream>>>(Ap, Bt, bias, res, C, M, Ncol, Kp, ldc,
                                              bTot, sR, sC, act, biasMode);
}

extern "C" void kernel_launch(void* const* d_in, const int* in_sizes, int n_in,
                              void* d_out, int out_size, void* d_ws, size_t ws_size,
                              hipStream_t stream)
{
    const float* x     = (const float*)d_in[0];
    const int*   esrc  = (const int*)d_in[1];
    const int*   edst  = (const int*)d_in[2];
    const int*   etyp  = (const int*)d_in[3];
    const float* Wq    = (const float*)d_in[4];
    const float* bq    = (const float*)d_in[5];
    const float* Wk    = (const float*)d_in[6];
    const float* bk    = (const float*)d_in[7];
    const float* Wv    = (const float*)d_in[8];
    const float* bv    = (const float*)d_in[9];
    const float* Wo    = (const float*)d_in[10];
    const float* bo    = (const float*)d_in[11];
    const float* ebias = (const float*)d_in[12];
    const float* g1    = (const float*)d_in[13];
    const float* beta1 = (const float*)d_in[14];
    const float* Wf1   = (const float*)d_in[15];
    const float* bf1   = (const float*)d_in[16];
    const float* Wf2   = (const float*)d_in[17];
    const float* bf2   = (const float*)d_in[18];
    const float* g2    = (const float*)d_in[19];
    const float* beta2 = (const float*)d_in[20];
    const float* rg    = (const float*)d_in[21];
    const float* rbeta = (const float*)d_in[22];
    const float* rW1   = (const float*)d_in[23];
    const float* rb1   = (const float*)d_in[24];
    const float* rW2   = (const float*)d_in[25];
    const float* rb2   = (const float*)d_in[26];
    const float* cg    = (const float*)d_in[27];
    const float* cbeta = (const float*)d_in[28];
    const float* cW1   = (const float*)d_in[29];
    const float* cb1   = (const float*)d_in[30];
    const float* cW2   = (const float*)d_in[31];
    const float* cb2   = (const float*)d_in[32];
    const float* mW0   = (const float*)d_in[33];
    const float* mb0   = (const float*)d_in[34];
    const float* mW1   = (const float*)d_in[35];
    const float* mb1   = (const float*)d_in[36];
    const float* mW2   = (const float*)d_in[37];
    const float* mb2   = (const float*)d_in[38];

    float* ws = (float*)d_ws;
    // workspace layout (floats), with heavy reuse after the GT stack
    float* h    = ws;                   // 10M  (N*D)
    float* q    = ws + 10000000LL;      // 10M  (later: ff1 20M / y / zb2)
    float* kk   = ws + 20000000LL;      // 10M  (later: zb / z12)
    float* v    = ws + 30000000LL;      // 10M  (later: z1 spans 20M / r2)
    float* agg  = ws + 40000000LL;      // 10M  (later: y3)
    float* t1   = ws + 50000000LL;      // 20M  (bn input / r)
    float* s    = ws + 70000000LL;      // 4M   (later: y2)
    float* mseg = ws + 74000000LL;      // 0.5M (later: p)
    float* zseg = ws + 74500000LL;      // 0.5M (later: o1/o2/logits)
    float* meanv= ws + 75000000LL;      // 512
    float* varv = ws + 75000512LL;      // 512
    float* pa   = ws + 75001024LL;      // 10.5M floats = bf16 pack A (max 50048x416)
    float* pb   = ws + 85501024LL;      // 11M   floats = bf16 pack Bt (max 50x1024x416)

    hipMemcpyAsync(h, x, sizeof(float) * (size_t)NN * DD,
                   hipMemcpyDeviceToDevice, stream);

    const int totEH = EE * HH;

    for (int l = 0; l < 2; ++l) {
        const float* Wq_l  = Wq  + (long long)l * DD * DD;
        const float* Wk_l  = Wk  + (long long)l * DD * DD;
        const float* Wv_l  = Wv  + (long long)l * DD * DD;
        const float* Wo_l  = Wo  + (long long)l * DD * DD;
        const float* Wf1_l = Wf1 + (long long)l * DD * 2 * DD;
        const float* Wf2_l = Wf2 + (long long)l * 2 * DD * DD;

        // Q/K/V projections (bf16 WMMA)
        run_gemm(stream, pa, pb, h, DD, Wq_l, DD, 0, bq + l * DD, nullptr, q,
                 NN, DD, DD, DD, 0, 0, 1, 0, 0);
        run_gemm(stream, pa, pb, h, DD, Wk_l, DD, 0, bk + l * DD, nullptr, kk,
                 NN, DD, DD, DD, 0, 0, 1, 0, 0);
        run_gemm(stream, pa, pb, h, DD, Wv_l, DD, 0, bv + l * DD, nullptr, v,
                 NN, DD, DD, DD, 0, 0, 1, 0, 0);

        // edge attention (segment softmax + scatter aggregate)
        attn_scores<<<gup(totEH), 256, 0, stream>>>(q, kk, esrc, edst, etyp,
                                                    ebias + l * ETc * HH, s, EE);
        fill_f32<<<gup((long long)NN * HH), 256, 0, stream>>>(mseg, -3.0e38f,
                                                              (long long)NN * HH);
        attn_max<<<gup(totEH), 256, 0, stream>>>(s, edst, mseg, EE);
        fill_f32<<<gup((long long)NN * HH), 256, 0, stream>>>(zseg, 0.0f,
                                                              (long long)NN * HH);
        attn_exp<<<gup(totEH), 256, 0, stream>>>(s, edst, mseg, zseg, EE);
        fill_f32<<<gup((long long)NN * DD), 256, 0, stream>>>(agg, 0.0f,
                                                              (long long)NN * DD);
        attn_agg<<<gup(totEH), 256, 0, stream>>>(s, v, esrc, edst, zseg, agg, EE);

        // output projection + residual, then BN -> h
        run_gemm(stream, pa, pb, agg, DD, Wo_l, DD, 0, bo + l * DD, h, t1,
                 NN, DD, DD, DD, 0, 0, 1, 0, 0);
        bn_stats<<<DD, 256, 0, stream>>>(t1, NN, DD, meanv, varv);
        bn_apply<<<gup((long long)NN * DD), 256, 0, stream>>>(
            t1, meanv, varv, g1 + l * DD, beta1 + l * DD, h, (long long)NN * DD, DD);

        // FFN: relu(h@Wf1+b) @ Wf2 + b + h, then BN -> h
        run_gemm(stream, pa, pb, h, DD, Wf1_l, 2 * DD, 0, bf1 + l * 2 * DD, nullptr, q,
                 NN, 2 * DD, DD, 2 * DD, 0, 0, 1, 1, 0);
        run_gemm(stream, pa, pb, q, 2 * DD, Wf2_l, DD, 0, bf2 + l * DD, h, t1,
                 NN, DD, 2 * DD, DD, 0, 0, 1, 0, 0);
        bn_stats<<<DD, 256, 0, stream>>>(t1, NN, DD, meanv, varv);
        bn_apply<<<gup((long long)NN * DD), 256, 0, stream>>>(
            t1, meanv, varv, g2 + l * DD, beta2 + l * DD, h, (long long)NN * DD, DD);
    }

    // de-batchify: y[b,c,l] (B x D x L1)
    float* y = q;
    to_bcl<<<gup((long long)BB * DD * L1c), 256, 0, stream>>>(h, y, BB * DD * L1c);

    // RepLK block: BN -> conv1x1 -> relu -> conv1x1, residual, relu, pool
    bnconv_stats<<<DD, 256, 0, stream>>>(y, BB, DD, L1c, meanv, varv);
    float* zb = kk;
    bnconv_apply<<<gup((long long)BB * DD * L1c), 256, 0, stream>>>(
        y, meanv, varv, rg, rbeta, zb, BB * DD * L1c, L1c, DD);
    float* z1 = v;  // B x 400 x L1 (spans 20M region)
    run_gemm(stream, pa, pb, rW1, DD, zb, L1c, (long long)DD * L1c, rb1, nullptr, z1,
             2 * DD, L1c, DD, L1c, 0, (long long)2 * DD * L1c, BB, 1, 1);
    float* r = t1;  // relu(y + conv)
    run_gemm(stream, pa, pb, rW2, 2 * DD, z1, L1c, (long long)2 * DD * L1c, rb2, y, r,
             DD, L1c, 2 * DD, L1c, (long long)DD * L1c, (long long)DD * L1c, BB, 1, 1);
    float* y2 = s;  // B x D x 333
    avgpool3<<<gup((long long)BB * DD * L2c), 256, 0, stream>>>(
        r, y2, L1c, L2c, BB * DD * L2c);

    // ConvFFN block: BN -> conv1x1 -> gelu -> conv1x1, residual, relu, pool
    bnconv_stats<<<DD, 256, 0, stream>>>(y2, BB, DD, L2c, meanv, varv);
    float* zb2 = q;
    bnconv_apply<<<gup((long long)BB * DD * L2c), 256, 0, stream>>>(
        y2, meanv, varv, cg, cbeta, zb2, BB * DD * L2c, L2c, DD);
    float* z12 = kk;  // B x 400 x 333
    run_gemm(stream, pa, pb, cW1, DD, zb2, L2c, (long long)DD * L2c, cb1, nullptr, z12,
             2 * DD, L2c, DD, L2c, 0, (long long)2 * DD * L2c, BB, 2, 1);
    float* r2 = v;
    run_gemm(stream, pa, pb, cW2, 2 * DD, z12, L2c, (long long)2 * DD * L2c, cb2, y2, r2,
             DD, L2c, 2 * DD, L2c, (long long)DD * L2c, (long long)DD * L2c, BB, 1, 1);
    float* y3 = agg;  // B x D x 111
    avgpool3<<<gup((long long)BB * DD * L3c), 256, 0, stream>>>(
        r2, y3, L2c, L3c, BB * DD * L3c);

    // readout: sum over length + MLP + softmax
    float* p = mseg;  // B x D
    rowsum<<<gup((long long)BB * DD), 256, 0, stream>>>(y3, p, L3c, BB * DD);
    float* o1 = zseg;              // 50 x 100
    float* o2 = zseg + 100000;     // 50 x 50
    float* lg = zseg + 200000;     // 50 x 2
    run_gemm(stream, pa, pb, p, DD, mW0, 100, 0, mb0, nullptr, o1,
             BB, 100, DD, 100, 0, 0, 1, 1, 0);
    run_gemm(stream, pa, pb, o1, 100, mW1, 50, 0, mb1, nullptr, o2,
             BB, 50, 100, 50, 0, 0, 1, 1, 0);
    run_gemm(stream, pa, pb, o2, 50, mW2, 2, 0, mb2, nullptr, lg,
             BB, 2, 50, 2, 0, 0, 1, 0, 0);
    softmax2<<<1, 64, 0, stream>>>(lg, (float*)d_out, BB);
}
